// VectorQuantizer_73753178407432
// MI455X (gfx1250) — matrix operand order repaired and checked by reading, not compiled, not run on hardware
//
#include <hip/hip_runtime.h>

// ---------------------------------------------------------------------------
// VQ-VAE vector quantizer for gfx1250 (MI455X).
// score[n,k] = |e_k|^2 - 2 * z_n . e_k   (|z|^2 dropped: constant per row)
// Score GEMM on v_wmma_f32_16x16x32_f16 (f32 accumulate); argmin in VALU with
// ds_swizzle cross-lane reduction; gather/losses/outputs in exact fp32.
// ---------------------------------------------------------------------------

typedef __attribute__((ext_vector_type(16))) _Float16 v16h;
typedef __attribute__((ext_vector_type(8)))  float    v8f;

#define KC     1024            // number of codes
#define DIM    64              // embedding dim
#define NT     (KC / 16)       // 64 N-tiles of 16 codes
#define BHW    (32 * 64 * 64)  // 131072 points
#define BDHW   (32 * 64 * 4096)
#define HWSTR  4096            // stride (floats) between dims d in z_e [B,D,H,W]
#define MT     4               // M-tiles (16 points each) per wave -> 64 points
#define WAVES  8               // waves per block -> 512 points per block

// ds_swizzle xor-exchange: new_lane = lane ^ OFF (and_mask=0x1f, or=0)
template <int OFF>
__device__ __forceinline__ float swz_f(float v) {
  constexpr int IMM = (OFF << 10) | 0x1F;
  return __int_as_float(__builtin_amdgcn_ds_swizzle(__float_as_int(v), IMM));
}
template <int OFF>
__device__ __forceinline__ int swz_i(int v) {
  constexpr int IMM = (OFF << 10) | 0x1F;
  return __builtin_amdgcn_ds_swizzle(v, IMM);
}

template <int OFF>
__device__ __forceinline__ void argmin_step(float (&v)[8], int (&ix)[8]) {
#pragma unroll
  for (int i = 0; i < 8; ++i) {
    const float ov = swz_f<OFF>(v[i]);
    const int   oi = swz_i<OFF>(ix[i]);
    if (ov < v[i] || (ov == v[i] && oi < ix[i])) { v[i] = ov; ix[i] = oi; }
  }
}

// ---------------------------------------------------------------------------
// Prep: pack B = (-2*E)^T into WMMA B-fragment layout (f16) and compute |e|^2.
// B fragment (32x16 f16, wave32): lane = 16*hi + col holds column n = col;
// element t of the v16h holds reduction-index Kb = t + 16*hi (this code's k order, same as A).
// Storage: eB[((nt*2 + kstep)*32 + lane)*16 + t]
// ---------------------------------------------------------------------------
__global__ void vq_prep(const float* __restrict__ emb,
                        _Float16* __restrict__ eB,
                        float* __restrict__ e2) {
  int tid = blockIdx.x * blockDim.x + threadIdx.x;
  if (tid < NT * 2 * 32) {
    int lane = tid & 31;
    int s    = (tid >> 5) & 1;   // kstep: d-range [32s, 32s+32)
    int nt   = tid >> 6;
    int n    = nt * 16 + (lane & 15);
    int hi   = lane >> 4;
    _Float16* dst = eB + (size_t)tid * 16;
#pragma unroll
    for (int t = 0; t < 16; ++t) {
      int d = 32 * s + t + 16 * hi;            // B-matrix K index -> dim d
      dst[t] = (_Float16)(-2.0f * emb[n * DIM + d]);
    }
  }
  if (tid < KC) {
    float acc = 0.0f;
#pragma unroll
    for (int d = 0; d < DIM; ++d) {
      float v = emb[tid * DIM + d];
      acc += v * v;
    }
    e2[tid] = acc;
  }
}

// ---------------------------------------------------------------------------
// Main: one wave per 64-point group (4 M-tiles, one full W-row);
// 8 waves (256 threads) per block.
// ---------------------------------------------------------------------------
__global__ __launch_bounds__(256) void vq_main(const float* __restrict__ z,
                                               const float* __restrict__ emb,
                                               const _Float16* __restrict__ eB,
                                               const float* __restrict__ e2,
                                               float* __restrict__ out_zq,
                                               float* __restrict__ out_idx,
                                               float* __restrict__ loss_accum) {
  __shared__ int s_idx[WAVES][MT][16];

  const int lane = threadIdx.x & 31;
  const int wave = threadIdx.x >> 5;
  const int hi   = lane >> 4;
  const int col  = lane & 15;

  const int grp  = blockIdx.x * WAVES + wave;   // 64-point group id
  const int base = grp * (MT * 16);             // multiple of 64 -> w0 == 0
  const int bb   = base >> 12;                  // batch
  const int hh   = (base >> 6) & 63;            // H coord

  // z element for (dim d, point-in-row m): z[zbase + d*HWSTR + m]
  const size_t zbase = (size_t)bb * 64 * HWSTR + (size_t)hh * 64;

  // --- load A fragments (16x32 f16 each): row m = col of tile mt ---------
  // A k order chosen here (matches B pack), element t: Ka = t + 16*hi
  v16h a0[MT], a1[MT];
#pragma unroll
  for (int mt = 0; mt < MT; ++mt) {
    const int m = mt * 16 + col;
#pragma unroll
    for (int t = 0; t < 16; ++t) {
      const int Ka = t + 16 * hi;
      a0[mt][t] = (_Float16)z[zbase + (size_t)Ka * HWSTR + m];
      a1[mt][t] = (_Float16)z[zbase + (size_t)(Ka + 32) * HWSTR + m];
    }
  }

  float minv[MT][8];
  int   mini[MT][8];
#pragma unroll
  for (int mt = 0; mt < MT; ++mt)
#pragma unroll
    for (int i = 0; i < 8; ++i) { minv[mt][i] = 3.0e38f; mini[mt][i] = 0; }

  // --- loop over 64 code tiles: 8 WMMAs per B-fragment load --------------
  for (int nt = 0; nt < NT; ++nt) {
    const v16h b0 = *(const v16h*)(eB + ((size_t)(nt * 2 + 0) * 32 + lane) * 16);
    const v16h b1 = *(const v16h*)(eB + ((size_t)(nt * 2 + 1) * 32 + lane) * 16);
    const float e2v  = e2[nt * 16 + col];       // per-column |e|^2
    const int   nidx = nt * 16 + col;
#pragma unroll
    for (int mt = 0; mt < MT; ++mt) {
      v8f c;
#pragma unroll
      for (int i = 0; i < 8; ++i) c[i] = e2v;
      c = __builtin_amdgcn_wmma_f32_16x16x32_f16(false, a0[mt], false, b0,
                                                 (short)0, c, false, false);
      c = __builtin_amdgcn_wmma_f32_16x16x32_f16(false, a1[mt], false, b1,
                                                 (short)0, c, false, false);
#pragma unroll
      for (int i = 0; i < 8; ++i) {
        if (c[i] < minv[mt][i]) { minv[mt][i] = c[i]; mini[mt][i] = nidx; }
      }
    }
  }

  // --- argmin across the 16 columns (xor within each 16-lane half) -------
#pragma unroll
  for (int mt = 0; mt < MT; ++mt) {
    argmin_step<8>(minv[mt], mini[mt]);
    argmin_step<4>(minv[mt], mini[mt]);
    argmin_step<2>(minv[mt], mini[mt]);
    argmin_step<1>(minv[mt], mini[mt]);
    // C layout: VGPR i, lanes 0-15 -> row i ; lanes 16-31 -> row i+8
    if (lane == 0) {
#pragma unroll
      for (int i = 0; i < 8; ++i) s_idx[wave][mt][i] = mini[mt][i];
    }
    if (lane == 16) {
#pragma unroll
      for (int i = 0; i < 8; ++i) s_idx[wave][mt][8 + i] = mini[mt][i];
    }
  }
  __syncthreads();

  // --- gather z_q (forward value of z_q_st), coalesced stores ------------
  // lanes 0-15: dims 0..31 ; lanes 16-31: dims 32..63; 16 consecutive w/half
  float acc = 0.0f;
#pragma unroll
  for (int mt = 0; mt < MT; ++mt) {
    const int row   = col;
    const int kbest = s_idx[wave][mt][row];
    if (hi == 0) out_idx[base + mt * 16 + row] = (float)kbest;
#pragma unroll
    for (int j = 0; j < 32; ++j) {
      const int d = hi * 32 + j;
      const size_t off = zbase + (size_t)d * HWSTR + mt * 16 + row;
      const float ev = emb[kbest * DIM + d];
      const float zv = z[off];
      out_zq[off] = ev;
      const float diff = ev - zv;
      acc += diff * diff;
    }
  }
  // wave-reduce squared error, one atomic per wave
  acc += swz_f<16>(acc);
  acc += swz_f<8>(acc);
  acc += swz_f<4>(acc);
  acc += swz_f<2>(acc);
  acc += swz_f<1>(acc);
  if (lane == 0) atomicAdd(loss_accum, acc);
}

// ---------------------------------------------------------------------------
__global__ void vq_finalize(const float* __restrict__ loss_accum,
                            float* __restrict__ out_losses) {
  const float mean = loss_accum[0] / (float)BDHW;
  out_losses[0] = mean;           // vq_loss
  out_losses[1] = 0.25f * mean;   // commitment_loss
}

// ---------------------------------------------------------------------------
extern "C" void kernel_launch(void* const* d_in, const int* in_sizes, int n_in,
                              void* d_out, int out_size, void* d_ws, size_t ws_size,
                              hipStream_t stream) {
  const float* z   = (const float*)d_in[0];   // [32,64,64,64]
  const float* emb = (const float*)d_in[1];   // [1024,64]

  float* out      = (float*)d_out;
  float* out_zq   = out;                      // 8388608 floats
  float* out_idx  = out + BDHW;               // 131072 floats
  float* out_loss = out + BDHW + BHW;         // 2 floats

  _Float16* eB  = (_Float16*)d_ws;                      // 128 KB packed B frags
  float*    e2  = (float*)((char*)d_ws + 128 * 1024);   // 4 KB |e|^2
  float*    acc = e2 + KC;                              // 4 B loss accumulator

  hipMemsetAsync(acc, 0, sizeof(float), stream);
  vq_prep<<<16, 256, 0, stream>>>(emb, eB, e2);
  vq_main<<<BHW / (WAVES * MT * 16), 256, 0, stream>>>(z, emb, eB, e2,
                                                       out_zq, out_idx, acc);
  vq_finalize<<<1, 1, 0, stream>>>(acc, out_loss);
}
